// HybridModel_16432544875302
// MI455X (gfx1250) — matrix operand order repaired and checked by reading, not compiled, not run on hardware
//
#include <hip/hip_runtime.h>
#include <hip/hip_bf16.h>

// ---------------------------------------------------------------------------
// HybridModel (text MLP + 2-layer GAT) for gfx1250.
// GEMMs: v_wmma_f32_16x16x32_f16, wave32, 16x(16*NT) strip per wave.
//  - A (activations, f32): 4x global_load_b128 + cvt_pk per k-step.
//  - B (weights): pre-packed once per call into f16 WMMA-fragment order
//    ([ktile][ntile][lane][16]), so each lane loads its fragment with 2
//    contiguous b128 loads - coalesced, no converts, no guards (zero-padded).
// Segment softmax over edges: ordered-uint atomicMax + f32 atomicAdd
// (entire GNN working set ~60MB is resident in the 192MB L2).
// ---------------------------------------------------------------------------

typedef __attribute__((ext_vector_type(16))) _Float16 v16h;
typedef __attribute__((ext_vector_type(8)))  float    v8f;

union AFrag { v16h v; _Float16 h[16]; };
union CFrag { v8f  v; float    f[8];  };

#define GDIM 128   // GNN feature dim
#define HID1 64    // per-head hidden of conv1
#define H1   2     // heads of conv1
#define OUTD 16    // output classes

// ---- order-preserving float<->uint mapping for atomicMax on f32 -----------
__device__ __forceinline__ unsigned fkey(float f) {
  unsigned u = __float_as_uint(f);
  return (u & 0x80000000u) ? ~u : (u | 0x80000000u);
}
__device__ __forceinline__ float fdec(unsigned k) {
  return __uint_as_float((k & 0x80000000u) ? (k ^ 0x80000000u) : ~k);
}

__global__ void k_zero(unsigned* p, int n) {
  int i = blockIdx.x * blockDim.x + threadIdx.x;
  if (i < n) p[i] = 0u;
}

// ---- pack f32 weight B[K,N] into f16 WMMA fragment order ------------------
// P[((kt*tilesN + nt)*32 + lane)*16 + j] = B[kt*32 + (lane>>4)*8 + (j&7) + (j>>3)*16][nt*16 + lane&15]
// Zero-padded at K/N edges so the GEMM needs no B-side guards.
__global__ void k_pack_b(const float* __restrict__ B, _Float16* __restrict__ P,
                         int Kt, int Nt, int tilesN, int total) {
  int p = blockIdx.x * blockDim.x + threadIdx.x;
  if (p >= total) return;
  int tile = p >> 9, r = p & 511;
  int lane = r >> 4, j = r & 15;
  int kt = tile / tilesN, nt = tile - kt * tilesN;
  int K = (kt << 5) + ((lane >> 4) << 3) + (j & 7) + ((j >> 3) << 4);
  int c = (nt << 4) + (lane & 15);
  P[p] = (K < Kt && c < Nt) ? (_Float16)B[(long)K * Nt + c] : (_Float16)0.f;
}

// ---------------------------------------------------------------------------
// C[M,N] = act( A[M,K] @ B[K,N] + bias + (accum ? C : 0) ), B pre-packed f16.
// Each wave owns NT consecutive 16x16 n-tiles of one m-tile: A fragment is
// loaded once per k-step and reused by NT WMMAs (independent accumulators).
// Caller guarantees tilesN % NT == 0. 8 waves per 256-thread block.
// Fragment layouts per CDNA5 ISA 7.12.2 (16-bit A 16x32, B 32x16, f32 C/D).
// ---------------------------------------------------------------------------
template <int NT>
__global__ void k_gemm(const float* __restrict__ A, const _Float16* __restrict__ Bp,
                       const float* __restrict__ bias, float* __restrict__ C,
                       int Mt, int Nt, int Kt, int relu, int accum) {
  int wave    = threadIdx.x >> 5;
  int lane    = threadIdx.x & 31;
  int tilesN  = (Nt + 15) >> 4;
  int tilesM  = (Mt + 15) >> 4;
  int groupsN = tilesN / NT;
  int group   = blockIdx.x * (blockDim.x >> 5) + wave;
  if (group >= tilesM * groupsN) return;         // wave-uniform: EXEC stays all-1s
  int m0 = (group / groupsN) << 4;
  int n0 = (group % groupsN) * (NT << 4);
  int lm = lane & 15;                            // A row / D col within tile
  int hi = lane >> 4;                            // lane group (0: lanes 0-15, 1: 16-31)

  CFrag c[NT];
  #pragma unroll
  for (int nt = 0; nt < NT; ++nt)
    #pragma unroll
    for (int j = 0; j < 8; ++j) c[nt].f[j] = 0.f;

  // Interior strip? (wave-uniform; K rows 16B-aligned when Kt%4==0)
  bool fullMN = (m0 + 16 <= Mt) && (n0 + (NT << 4) <= Nt) && ((Kt & 3) == 0);
  int  kfull  = fullMN ? (Kt & ~31) : 0;

  const float*    Arow  = A + (long)(m0 + lm) * Kt + (hi << 3);
  const _Float16* Bbase = Bp + ((long)(n0 >> 4) << 9) + (lane << 4);

  for (int k0 = 0; k0 < kfull; k0 += 32) {
    const float*    Ap = Arow + k0;
    const _Float16* Bk = Bbase + (((long)(k0 >> 5) * tilesN) << 9);
    __builtin_prefetch(Ap + 32, 0, 3);           // global_prefetch_b8 (next k-step)
    float4 a0 = *(const float4*)(Ap);
    float4 a1 = *(const float4*)(Ap + 4);
    float4 a2 = *(const float4*)(Ap + 16);
    float4 a3 = *(const float4*)(Ap + 20);
    AFrag a, b[NT];
    #pragma unroll
    for (int nt = 0; nt < NT; ++nt)
      b[nt].v = *(const v16h*)(Bk + ((long)nt << 9));    // 2x b128, coalesced
    a.h[0]=(_Float16)a0.x; a.h[1]=(_Float16)a0.y; a.h[2]=(_Float16)a0.z; a.h[3]=(_Float16)a0.w;
    a.h[4]=(_Float16)a1.x; a.h[5]=(_Float16)a1.y; a.h[6]=(_Float16)a1.z; a.h[7]=(_Float16)a1.w;
    a.h[8]=(_Float16)a2.x; a.h[9]=(_Float16)a2.y; a.h[10]=(_Float16)a2.z; a.h[11]=(_Float16)a2.w;
    a.h[12]=(_Float16)a3.x; a.h[13]=(_Float16)a3.y; a.h[14]=(_Float16)a3.z; a.h[15]=(_Float16)a3.w;
    #pragma unroll
    for (int nt = 0; nt < NT; ++nt)
      c[nt].v = __builtin_amdgcn_wmma_f32_16x16x32_f16(false, a.v, false, b[nt].v,
                                                       (short)0, c[nt].v, false, false);
  }

  // Guarded K-tail / boundary strips (A-side guards only; B pack is padded).
  for (int k0 = kfull; k0 < Kt; k0 += 32) {
    const _Float16* Bk = Bbase + (((long)(k0 >> 5) * tilesN) << 9);
    AFrag a;
    #pragma unroll
    for (int j = 0; j < 16; ++j) {
      int v  = j >> 1, hb = j & 1;
      int ka = (v < 4 ? 2 * v + hb : 16 + 2 * (v - 4) + hb) + (hi << 3);
      int ar = m0 + lm, ac = k0 + ka;
      a.h[j] = (ar < Mt && ac < Kt) ? (_Float16)A[(long)ar * Kt + ac] : (_Float16)0.f;
    }
    #pragma unroll
    for (int nt = 0; nt < NT; ++nt) {
      AFrag b;
      b.v = *(const v16h*)(Bk + ((long)nt << 9));
      c[nt].v = __builtin_amdgcn_wmma_f32_16x16x32_f16(false, a.v, false, b.v,
                                                       (short)0, c[nt].v, false, false);
    }
  }

  // D layout: VGPR j -> row (j + 8*hi), col = lane%16
  if (fullMN) {
    #pragma unroll
    for (int nt = 0; nt < NT; ++nt) {
      int col = n0 + (nt << 4) + lm;
      #pragma unroll
      for (int j = 0; j < 8; ++j) {
        long idx = (long)(m0 + j + (hi << 3)) * Nt + col;
        float v = c[nt].f[j];
        if (bias)  v += bias[col];
        if (accum) v += C[idx];
        if (relu)  v = v > 0.f ? v : 0.f;
        C[idx] = v;
      }
    }
  } else {
    #pragma unroll
    for (int nt = 0; nt < NT; ++nt) {
      int col = n0 + (nt << 4) + lm;
      #pragma unroll
      for (int j = 0; j < 8; ++j) {
        int r = m0 + j + (hi << 3);
        if (r < Mt && col < Nt) {
          float v = c[nt].f[j];
          if (bias)  v += bias[col];
          if (accum) v += C[(long)r * Nt + col];
          if (relu)  v = v > 0.f ? v : 0.f;
          C[(long)r * Nt + col] = v;
        }
      }
    }
  }
}

// ---- attention logits: a_s[n,h] = sum_d h1[n,h,d]*att_src[h,d] ------------
__global__ void k_attn1(const float* __restrict__ h1, const float* __restrict__ asrc,
                        const float* __restrict__ adst, float* a_s, float* a_d) {
  __shared__ float ss[128], sd[128];
  int n = blockIdx.x, t = threadIdx.x;           // 128 threads = one row
  int hh = t >> 6, d = t & 63;
  float v = h1[(long)n * GDIM + t];              // row layout [H][64] == att flat index
  ss[t] = v * asrc[t];
  sd[t] = v * adst[t];
  __syncthreads();
  for (int s = 32; s > 0; s >>= 1) {
    if (d < s) { ss[t] += ss[t + s]; sd[t] += sd[t + s]; }
    __syncthreads();
  }
  if (d == 0) { a_s[n * 2 + hh] = ss[t]; a_d[n * 2 + hh] = sd[t]; }
}

__global__ void k_attn2(const float* __restrict__ h2, const float* __restrict__ asrc,
                        const float* __restrict__ adst, float* a_s, float* a_d, int N) {
  int n = blockIdx.x * blockDim.x + threadIdx.x;
  if (n >= N) return;
  float s = 0.f, dd = 0.f;
  #pragma unroll
  for (int i = 0; i < OUTD; ++i) {
    float v = h2[(long)n * OUTD + i];
    s += v * asrc[i]; dd += v * adst[i];
  }
  a_s[n] = s; a_d[n] = dd;
}

// ---- edge pass 1: e = leaky_relu(a_s[src]+a_d[dst]); segment max ----------
__global__ void k_edge_max(const int* __restrict__ src, const int* __restrict__ dst,
                           int E, int N, int H, const float* __restrict__ a_s,
                           const float* __restrict__ a_d,
                           float* __restrict__ e_out, unsigned* __restrict__ emax) {
  int t = blockIdx.x * blockDim.x + threadIdx.x;
  int total = (E + N) * H;
  if (t >= total) return;
  int idx = t / H, h = t - idx * H;
  int s = idx < E ? src[idx] : idx - E;          // [E, E+N): self loop
  int d = idx < E ? dst[idx] : idx - E;
  float e = a_s[s * H + h] + a_d[d * H + h];
  e = e > 0.f ? e : 0.2f * e;                    // leaky_relu(0.2)
  e_out[t] = e;
  atomicMax(&emax[d * H + h], fkey(e));
}

// ---- edge pass 2: ex = exp(e - max[dst]); segment sum ---------------------
__global__ void k_edge_exp(const int* __restrict__ src, const int* __restrict__ dst,
                           int E, int N, int H, float* __restrict__ e_out,
                           const unsigned* __restrict__ emax, float* __restrict__ den) {
  int t = blockIdx.x * blockDim.x + threadIdx.x;
  int total = (E + N) * H;
  if (t >= total) return;
  int idx = t / H, h = t - idx * H;
  int d = idx < E ? dst[idx] : idx - E;
  float ex = expf(e_out[t] - fdec(emax[d * H + h]));
  e_out[t] = ex;
  atomicAdd(&den[d * H + h], ex);
}

// ---- edge pass 3: acc[dst] += alpha * h[src] (float4 chunks) --------------
__global__ void k_edge_scatter(const int* __restrict__ src, const int* __restrict__ dst,
                               int E, int N, int H, int D,
                               const float* __restrict__ e_out, const float* __restrict__ den,
                               const float* __restrict__ hsrc, float* __restrict__ acc) {
  int chunks = (H * D) >> 2;                     // float4 chunks per edge
  int t = blockIdx.x * blockDim.x + threadIdx.x;
  int total = (E + N) * chunks;
  if (t >= total) return;
  int idx = t / chunks, c = t - idx * chunks;
  int h = c / (D >> 2);
  int q = (c - h * (D >> 2)) << 2;
  int s = idx < E ? src[idx] : idx - E;
  int d = idx < E ? dst[idx] : idx - E;
  float alpha = e_out[idx * H + h] / (den[d * H + h] + 1e-16f);
  const float4 hv = *(const float4*)&hsrc[(long)s * H * D + h * D + q];
  float* ap = &acc[(long)d * H * D + h * D + q];
  atomicAdd(ap + 0, hv.x * alpha);
  atomicAdd(ap + 1, hv.y * alpha);
  atomicAdd(ap + 2, hv.z * alpha);
  atomicAdd(ap + 3, hv.w * alpha);
}

__global__ void k_bias_act(float* buf, const float* __restrict__ bias,
                           int total, int cols, int relu) {
  int i = blockIdx.x * blockDim.x + threadIdx.x;
  if (i >= total) return;
  float v = buf[i] + bias[i % cols];
  if (relu) v = v > 0.f ? v : 0.f;
  buf[i] = v;
}

__global__ void k_concat(const float* __restrict__ text_out, const float* __restrict__ out2,
                         const int* __restrict__ mask, float* __restrict__ comb, int M) {
  int t = blockIdx.x * blockDim.x + threadIdx.x;
  if (t >= M * 32) return;
  int m = t >> 5, c = t & 31;
  comb[t] = (c < OUTD) ? text_out[(long)m * OUTD + c]
                       : out2[(long)mask[m] * OUTD + (c - OUTD)];
}

__global__ void k_logsoftmax(float* out, int M) {
  int m = blockIdx.x * blockDim.x + threadIdx.x;
  if (m >= M) return;
  float* row = out + (long)m * OUTD;
  float mx = row[0];
  #pragma unroll
  for (int i = 1; i < OUTD; ++i) mx = fmaxf(mx, row[i]);
  float s = 0.f;
  #pragma unroll
  for (int i = 0; i < OUTD; ++i) s += expf(row[i] - mx);
  float l = mx + logf(s);
  #pragma unroll
  for (int i = 0; i < OUTD; ++i) row[i] = row[i] - l;
}

// ---------------------------------------------------------------------------
extern "C" void kernel_launch(void* const* d_in, const int* in_sizes, int n_in,
                              void* d_out, int out_size, void* d_ws, size_t ws_size,
                              hipStream_t stream) {
  const float* text  = (const float*)d_in[0];
  const float* x     = (const float*)d_in[1];
  const int*   ei    = (const int*)  d_in[2];
  const int*   mask  = (const int*)  d_in[3];
  const float* fc1_w = (const float*)d_in[4],  *fc1_b = (const float*)d_in[5];
  const float* fc2_w = (const float*)d_in[6],  *fc2_b = (const float*)d_in[7];
  const float* fc3_w = (const float*)d_in[8],  *fc3_b = (const float*)d_in[9];
  const float* fc4_w = (const float*)d_in[10], *fc4_b = (const float*)d_in[11];
  const float* sc_w  = (const float*)d_in[12], *sc_b  = (const float*)d_in[13];
  const float* W1    = (const float*)d_in[14];
  const float* as1   = (const float*)d_in[15], *ad1 = (const float*)d_in[16];
  const float* b1    = (const float*)d_in[17];
  const float* W2    = (const float*)d_in[18];
  const float* as2   = (const float*)d_in[19], *ad2 = (const float*)d_in[20];
  const float* b2    = (const float*)d_in[21];
  const float* fcf_w = (const float*)d_in[22], *fcf_b = (const float*)d_in[23];

  const int M    = in_sizes[3];
  const int N    = in_sizes[1] / GDIM;
  const int E    = in_sizes[2] / 2;
  const int TEXT = in_sizes[0] / M;
  const int* srcI = ei;
  const int* dstI = ei + E;

  // ---- workspace carve-out (element offsets, 1KB aligned for float4) ----
  float* ws = (float*)d_ws;
  size_t off = 0;
  auto alloc = [&](size_t n) { size_t o = off; off += (n + 255) & ~(size_t)255; return o; };
  size_t o_emax1 = alloc((size_t)2 * N);         // zeroed region begins
  size_t o_den1  = alloc((size_t)2 * N);
  size_t o_acc1  = alloc((size_t)GDIM * N);      // becomes g after bias+relu
  size_t o_emax2 = alloc((size_t)N);
  size_t o_den2  = alloc((size_t)N);
  size_t o_acc2  = alloc((size_t)OUTD * N);      // becomes out2 after bias
  size_t zero_end = off;                         // zeroed region ends
  size_t o_h1  = alloc((size_t)GDIM * N);
  size_t o_as1 = alloc((size_t)2 * N);
  size_t o_ad1 = alloc((size_t)2 * N);
  size_t o_e1  = alloc((size_t)2 * (E + N));
  size_t o_h2  = alloc((size_t)OUTD * N);
  size_t o_s2  = alloc((size_t)N);
  size_t o_d2  = alloc((size_t)N);
  size_t o_e2  = alloc((size_t)(E + N));
  size_t o_t1  = alloc((size_t)128 * M);
  size_t o_t2  = alloc((size_t)64 * M);
  size_t o_t3  = alloc((size_t)OUTD * M);
  size_t o_to  = alloc((size_t)OUTD * M);
  size_t o_cb  = alloc((size_t)2 * OUTD * M);
  size_t o_wp  = alloc((size_t)65536);           // 256KB: packed f16 weights
  (void)ws_size; (void)n_in; (void)out_size;

  auto cdiv = [](int a, int b) { return (a + b - 1) / b; };

  // ---- pack all weight matrices to f16 fragment order (tiny, per call) ----
  _Float16* wp = (_Float16*)(ws + o_wp);
  size_t hoff = 0;
  auto packB = [&](const float* B, int Kt, int Nt) {
    int tK = (Kt + 31) / 32, tN = (Nt + 15) / 16;
    int n = tK * tN * 512;
    _Float16* P = wp + hoff;
    hoff += (size_t)n;
    k_pack_b<<<cdiv(n, 256), 256, 0, stream>>>(B, P, Kt, Nt, tN, n);
    return (const _Float16*)P;
  };
  const _Float16* pW1  = packB(W1,    GDIM, GDIM);
  const _Float16* pW2  = packB(W2,    GDIM, OUTD);
  const _Float16* pF1  = packB(fc1_w, TEXT, 128);
  const _Float16* pF2  = packB(fc2_w, 128,  64);
  const _Float16* pF3  = packB(fc3_w, 64,   OUTD);
  const _Float16* pF4  = packB(fc4_w, OUTD, OUTD);
  const _Float16* pSC  = packB(sc_w,  TEXT, OUTD);
  const _Float16* pFCF = packB(fcf_w, 2 * OUTD, OUTD);

  // NT=4 strip kernel when tilesN divisible by 4, else single-tile kernel.
  auto gemm = [&](const float* A, const _Float16* Bp, const float* bias, float* C,
                  int Mt, int Nt, int Kt, int relu, int accum) {
    int tilesM = (Mt + 15) / 16, tilesN = (Nt + 15) / 16;
    if ((tilesN & 3) == 0) {
      int groups = tilesM * (tilesN / 4);
      k_gemm<4><<<(groups + 7) / 8, 256, 0, stream>>>(A, Bp, bias, C, Mt, Nt, Kt, relu, accum);
    } else {
      int groups = tilesM * tilesN;
      k_gemm<1><<<(groups + 7) / 8, 256, 0, stream>>>(A, Bp, bias, C, Mt, Nt, Kt, relu, accum);
    }
  };

  // 0) zero accumulators / softmax stats
  k_zero<<<cdiv((int)zero_end, 256), 256, 0, stream>>>((unsigned*)d_ws, (int)zero_end);

  // ---- GAT layer 1 ----
  gemm(x, pW1, nullptr, ws + o_h1, N, GDIM, GDIM, 0, 0);                // h1 = x@W1
  k_attn1<<<N, 128, 0, stream>>>(ws + o_h1, as1, ad1, ws + o_as1, ws + o_ad1);
  {
    int tot = (E + N) * H1;
    k_edge_max<<<cdiv(tot, 256), 256, 0, stream>>>(srcI, dstI, E, N, H1,
        ws + o_as1, ws + o_ad1, ws + o_e1, (unsigned*)(ws + o_emax1));
    k_edge_exp<<<cdiv(tot, 256), 256, 0, stream>>>(srcI, dstI, E, N, H1,
        ws + o_e1, (const unsigned*)(ws + o_emax1), ws + o_den1);
    int tots = (E + N) * ((H1 * HID1) >> 2);
    k_edge_scatter<<<cdiv(tots, 256), 256, 0, stream>>>(srcI, dstI, E, N, H1, HID1,
        ws + o_e1, ws + o_den1, ws + o_h1, ws + o_acc1);
  }
  k_bias_act<<<cdiv(N * GDIM, 256), 256, 0, stream>>>(ws + o_acc1, b1, N * GDIM, GDIM, 1);

  // ---- GAT layer 2 (1 head, dim 16, concat=False -> identity mean) ----
  gemm(ws + o_acc1, pW2, nullptr, ws + o_h2, N, OUTD, GDIM, 0, 0);      // h2 = g@W2
  k_attn2<<<cdiv(N, 256), 256, 0, stream>>>(ws + o_h2, as2, ad2, ws + o_s2, ws + o_d2, N);
  {
    int tot = (E + N);
    k_edge_max<<<cdiv(tot, 256), 256, 0, stream>>>(srcI, dstI, E, N, 1,
        ws + o_s2, ws + o_d2, ws + o_e2, (unsigned*)(ws + o_emax2));
    k_edge_exp<<<cdiv(tot, 256), 256, 0, stream>>>(srcI, dstI, E, N, 1,
        ws + o_e2, (const unsigned*)(ws + o_emax2), ws + o_den2);
    int tots = (E + N) * (OUTD >> 2);
    k_edge_scatter<<<cdiv(tots, 256), 256, 0, stream>>>(srcI, dstI, E, N, 1, OUTD,
        ws + o_e2, ws + o_den2, ws + o_h2, ws + o_acc2);
  }
  k_bias_act<<<cdiv(N * OUTD, 256), 256, 0, stream>>>(ws + o_acc2, b2, N * OUTD, OUTD, 0);

  // ---- text MLP branch + shortcut ----
  gemm(text, pF1, fc1_b, ws + o_t1, M, 128, TEXT, 1, 0);
  gemm(ws + o_t1, pF2, fc2_b, ws + o_t2, M, 64, 128, 1, 0);
  gemm(ws + o_t2, pF3, fc3_b, ws + o_t3, M, OUTD, 64, 1, 0);
  gemm(ws + o_t3, pF4, fc4_b, ws + o_to, M, OUTD, OUTD, 0, 0);
  gemm(text, pSC, sc_b, ws + o_to, M, OUTD, TEXT, 0, 1);                // += shortcut

  // ---- combine, classify, log_softmax ----
  k_concat<<<cdiv(M * 32, 256), 256, 0, stream>>>(ws + o_to, ws + o_acc2, mask,
                                                  ws + o_cb, M);
  gemm(ws + o_cb, pFCF, fcf_b, (float*)d_out, M, OUTD, 2 * OUTD, 0, 0);
  k_logsoftmax<<<cdiv(M, 256), 256, 0, stream>>>((float*)d_out, M);
}
